// BiLSTMLayer_14877766713393
// MI455X (gfx1250) — matrix operand order repaired and checked by reading, not compiled, not run on hardware
//
#include <hip/hip_runtime.h>
#include <cstdint>
#include <cstddef>

// ---------------------------------------------------------------------------
// BiLSTM (2 layers, bidirectional), B=32, S=512, H=512, fused-gate WMMA scan.
// Per (layer,direction): one persistent 1024-thread workgroup runs the scan;
// each step computes gates = [x_t ; h] * [Wih ; Whh]^T with
// v_wmma_f32_16x16x32_bf16 (fp32 accumulate), keeps h (bf16) and c (f32) in
// LDS, stages x_t via global_load_async_to_lds_b128 (ASYNCcnt), and does the
// LSTM cell update with branch-free v_exp/v_rcp activations.
// ---------------------------------------------------------------------------

typedef __bf16 bf16;
typedef bf16  v16bf __attribute__((ext_vector_type(16)));
typedef float v8f   __attribute__((ext_vector_type(8)));

#define BATCH 32
#define SEQ   512
#define HID   512
#define GATES 2048   // 4*H

union FragAB {
  v16bf v;
  uint4 q[2];
};

__device__ inline v8f wmma_bf16(v16bf a, v16bf b, v8f c) {
  // D = A(16x32) * B(32x16) + C, fp32 accumulate
  return __builtin_amdgcn_wmma_f32_16x16x32_bf16(
      /*neg_a=*/false, a, /*neg_b=*/false, b,
      /*c_mod=*/(short)0, c, /*reuse_a=*/false, /*reuse_b=*/false);
}

// Branch-free activations: v_exp_f32 + v_rcp_f32, saturate correctly at +-inf.
__device__ inline float fast_sigmoid(float x) {
  return __builtin_amdgcn_rcpf(1.0f + __expf(-x));
}
__device__ inline float fast_tanh(float x) {
  return 1.0f - 2.0f * __builtin_amdgcn_rcpf(1.0f + __expf(2.0f * x));
}

// A fragment (16x32 bf16) from the LDS activation buffer [32][KA].
// ISA 7.12.2: lanes 0-15 hold rows M=0..15 K={0..7,16..23}; lanes 16-31 same
// rows, K={8..15,24..31}.  -> two 16B ds loads per lane.
__device__ inline v16bf load_a_frag(const bf16* a_lds, int KA, int mrow0,
                                    int k0, int lane) {
  int row = mrow0 + (lane & 15);
  int khi = (lane >> 4) * 8;
  const bf16* p = a_lds + row * KA + k0 + khi;
  FragAB f;
  f.q[0] = *(const uint4*)(p);       // K = khi .. khi+7
  f.q[1] = *(const uint4*)(p + 16);  // K = khi+16 .. khi+23
  return f.v;
}

// B fragment (32x16 bf16): B[k][n] = W[n][k], W row-major (GATES, ldw).
// Lanes 0-15: n=lane, K={0..7,16..23}; lanes 16-31: n=lane-16, K={8..15,24..31}.
// -> two 8-element bf16 runs along k per lane = two 16B global loads.
__device__ inline v16bf load_b_frag(const bf16* W, int ldw, int n0, int k0,
                                    int lane) {
  int n  = n0 + (lane & 15);
  int kb = k0 + ((lane >> 4) << 3);
  const bf16* p = W + (size_t)n * (size_t)ldw + kb;
  FragAB f;
  f.q[0] = *(const uint4*)(p);
  f.q[1] = *(const uint4*)(p + 16);
  return f.v;
}

// ---------------------------------------------------------------------------
// Conversion kernels (fp32 -> bf16)
// ---------------------------------------------------------------------------
__global__ void cvt_flat_kernel(const float* __restrict__ in,
                                bf16* __restrict__ out, int n) {
  for (int i = blockIdx.x * blockDim.x + threadIdx.x; i < n;
       i += gridDim.x * blockDim.x)
    out[i] = (bf16)in[i];
}

// x: (B,S,D) f32  ->  xbf: (S,B,D) bf16  (time-major so each step's A-matrix
// block of 32 rows is one contiguous chunk)
__global__ void cvt_x_kernel(const float* __restrict__ x,
                             bf16* __restrict__ out) {
  const int total = BATCH * SEQ * 512;
  for (int e = blockIdx.x * blockDim.x + threadIdx.x; e < total;
       e += gridDim.x * blockDim.x) {
    int s  = e >> 14;          // 32*512 = 16384 elems per timestep
    int b  = (e >> 9) & 31;
    int dd = e & 511;
    out[e] = (bf16)x[((size_t)b * SEQ + s) * 512 + dd];
  }
}

// ---------------------------------------------------------------------------
// Persistent LSTM scan kernel: grid = 2 blocks (dir 0 = fwd, 1 = bwd),
// block = 1024 threads = 32 waves.
//   wave w: mt = w&1 (batch tile), jt0 = w>>1; tasks jt in {jt0, jt0+16}
//   each task owns hidden cols [jt*16, jt*16+16) for ALL four gates.
// LDS layout (dynamic): a_lds[32][KA] bf16 (KA = IN_DIM+512; x_t then h),
//                       c_lds[32][512] f32.
// ---------------------------------------------------------------------------
template <int IN_DIM, bool FINAL>
__global__ __launch_bounds__(1024) void lstm_scan_kernel(
    const bf16* __restrict__ xin,   // (S, 32, IN_DIM) bf16
    const bf16* __restrict__ wih_f, const float* __restrict__ bias_f,
    const bf16* __restrict__ whh_f,
    const bf16* __restrict__ wih_b, const float* __restrict__ bias_b,
    const bf16* __restrict__ whh_b,
    bf16* __restrict__ ybf,         // (S, 32, 1024) bf16   (used if !FINAL)
    float* __restrict__ yout)       // (32, 512, 1024) f32  (used if FINAL)
{
  constexpr int KA = IN_DIM + HID;
  extern __shared__ char smem[];
  bf16*  a_lds = (bf16*)smem;                                   // 32*KA bf16
  float* c_lds = (float*)(smem + (size_t)BATCH * KA * sizeof(bf16));

  const int dir  = blockIdx.x;
  const int tid  = threadIdx.x;
  const int lane = tid & 31;
  const int w    = tid >> 5;
  const int mt   = w & 1;          // batch tile 0/1
  const int jt0  = w >> 1;         // hidden tile base 0..15
  const int mrow0 = mt * 16;
  const int hioff = (lane >> 4) << 3;   // +8 rows for high half-wave

  const bf16*  Wih  = dir ? wih_b  : wih_f;
  const bf16*  Whh  = dir ? whh_b  : whh_f;
  const float* bias = dir ? bias_b : bias_f;

  // loop-invariant per-lane biases (i,f,g,o per task)
  float bias_r[2][4];
#pragma unroll
  for (int task = 0; task < 2; ++task) {
    const int col = (jt0 + task * 16) * 16 + (lane & 15);
#pragma unroll
    for (int q = 0; q < 4; ++q) bias_r[task][q] = bias[q * HID + col];
  }

  // ---- init: c = 0, h = 0 --------------------------------------------------
  {
    uint4 z4; z4.x = z4.y = z4.z = z4.w = 0u;
    float4 zf; zf.x = zf.y = zf.z = zf.w = 0.0f;
    for (int ci = tid; ci < (BATCH * HID) / 4; ci += 1024)
      ((float4*)c_lds)[ci] = zf;
    for (int ci = tid; ci < BATCH * (HID / 8); ci += 1024) {
      int b = ci >> 6, off = ci & 63;
      *(uint4*)(a_lds + (size_t)b * KA + IN_DIM + off * 8) = z4;
    }
  }

  constexpr int CPR = IN_DIM / 8;          // 16B chunks per row
  constexpr int CH  = BATCH * CPR;         // chunks per timestep

  for (int t = 0; t < SEQ; ++t) {
    const int s = dir ? (SEQ - 1 - t) : t;

    // ---- stage x_t into a_lds[:, 0:IN_DIM] via async global->LDS DMA ------
    {
      const bf16* src = xin + (size_t)s * BATCH * IN_DIM;
      for (int ci = tid; ci < CH; ci += 1024) {
        int b = ci / CPR, off = ci % CPR;
        unsigned lds_byte =
            (unsigned)((size_t)b * KA * sizeof(bf16) + (size_t)off * 16);
        unsigned long long gaddr =
            (unsigned long long)(uintptr_t)(const void*)(src + ci * 8);
        asm volatile("global_load_async_to_lds_b128 %0, %1, off"
                     :
                     : "v"(lds_byte), "v"(gaddr)
                     : "memory");
      }
      asm volatile("s_wait_asynccnt 0x0" ::: "memory");
    }
    // prefetch next timestep's x block (global_prefetch_b8)
    {
      int sn = dir ? (s - 1) : (s + 1);
      if (sn >= 0 && sn < SEQ) {
        const bf16* pn =
            xin + (size_t)sn * BATCH * IN_DIM +
            (size_t)tid * (BATCH * IN_DIM / 1024);
        __builtin_prefetch(pn, 0, 1);
      }
    }
    __syncthreads();   // x staged, h from previous step visible

    float hv[2][8];

    // ---- gates GEMM + cell update, one hidden tile per task ---------------
    for (int task = 0; task < 2; ++task) {
      const int jt  = jt0 + task * 16;
      const int col = jt * 16 + (lane & 15);   // hidden unit j

      v8f acc[4] = {};                          // i, f, g, o accumulators
      for (int kt = 0; kt < KA / 32; ++kt) {
        const int k0 = kt * 32;
        v16bf af = load_a_frag(a_lds, KA, mrow0, k0, lane);
        const bf16* W;
        int ldw, kk;
        if (k0 < IN_DIM) { W = Wih; ldw = IN_DIM; kk = k0; }
        else             { W = Whh; ldw = HID;    kk = k0 - IN_DIM; }
#pragma unroll
        for (int q = 0; q < 4; ++q) {
          v16bf bfrag = load_b_frag(W, ldw, q * HID + jt * 16, kk, lane);
          acc[q] = wmma_bf16(af, bfrag, acc[q]);
        }
      }

#pragma unroll
      for (int r = 0; r < 8; ++r) {
        const int brow = mrow0 + r + hioff;     // batch row
        const float iv = fast_sigmoid(acc[0][r] + bias_r[task][0]);
        const float fv = fast_sigmoid(acc[1][r] + bias_r[task][1]);
        const float gv = fast_tanh(acc[2][r] + bias_r[task][2]);
        const float ov = fast_sigmoid(acc[3][r] + bias_r[task][3]);
        const int cidx = brow * HID + col;
        const float cn = fv * c_lds[cidx] + iv * gv;
        c_lds[cidx] = cn;
        const float h = ov * fast_tanh(cn);
        hv[task][r] = h;
        if (FINAL) {
          yout[((size_t)brow * SEQ + s) * 1024 + dir * HID + col] = h;
        } else {
          ybf[((size_t)s * BATCH + brow) * 1024 + dir * HID + col] = (bf16)h;
        }
      }
    }

    __syncthreads();   // all reads of h region complete before overwrite

    // ---- publish new h into a_lds[:, IN_DIM:IN_DIM+512] -------------------
#pragma unroll
    for (int task = 0; task < 2; ++task) {
      const int jt  = jt0 + task * 16;
      const int col = jt * 16 + (lane & 15);
#pragma unroll
      for (int r = 0; r < 8; ++r) {
        const int brow = mrow0 + r + hioff;
        a_lds[(size_t)brow * KA + IN_DIM + col] = (bf16)hv[task][r];
      }
    }
    // next iteration's x-staging touches a disjoint LDS region; its
    // __syncthreads() orders these h writes before the next GEMM reads.
  }
}

// ---------------------------------------------------------------------------
// Host launcher
// ---------------------------------------------------------------------------
extern "C" void kernel_launch(void* const* d_in, const int* in_sizes, int n_in,
                              void* d_out, int out_size, void* d_ws,
                              size_t ws_size, hipStream_t stream) {
  (void)in_sizes; (void)n_in; (void)out_size; (void)ws_size;

  const float* x       = (const float*)d_in[0];
  const float* w_ih_f0 = (const float*)d_in[1];
  const float* b_ih_f0 = (const float*)d_in[2];
  const float* w_hh_f0 = (const float*)d_in[3];
  const float* w_ih_b0 = (const float*)d_in[4];
  const float* b_ih_b0 = (const float*)d_in[5];
  const float* w_hh_b0 = (const float*)d_in[6];
  const float* w_ih_f1 = (const float*)d_in[7];
  const float* b_ih_f1 = (const float*)d_in[8];
  const float* w_hh_f1 = (const float*)d_in[9];
  const float* w_ih_b1 = (const float*)d_in[10];
  const float* b_ih_b1 = (const float*)d_in[11];
  const float* w_hh_b1 = (const float*)d_in[12];

  // workspace layout (bf16 copies)
  char* ws = (char*)d_ws;
  bf16* xbf = (bf16*)ws;                                   // 512*32*512
  bf16* y0  = (bf16*)(ws + (size_t)16 * 1024 * 1024);      // 512*32*1024
  bf16* wp  = (bf16*)(ws + (size_t)48 * 1024 * 1024);
  bf16* wihF0 = wp;                 wp += (size_t)GATES * 512;
  bf16* whhF0 = wp;                 wp += (size_t)GATES * 512;
  bf16* wihB0 = wp;                 wp += (size_t)GATES * 512;
  bf16* whhB0 = wp;                 wp += (size_t)GATES * 512;
  bf16* wihF1 = wp;                 wp += (size_t)GATES * 1024;
  bf16* whhF1 = wp;                 wp += (size_t)GATES * 512;
  bf16* wihB1 = wp;                 wp += (size_t)GATES * 1024;
  bf16* whhB1 = wp;                 wp += (size_t)GATES * 512;

  const int thr = 256;
  cvt_x_kernel<<<2048, thr, 0, stream>>>(x, xbf);
  cvt_flat_kernel<<<1024, thr, 0, stream>>>(w_ih_f0, wihF0, GATES * 512);
  cvt_flat_kernel<<<1024, thr, 0, stream>>>(w_hh_f0, whhF0, GATES * 512);
  cvt_flat_kernel<<<1024, thr, 0, stream>>>(w_ih_b0, wihB0, GATES * 512);
  cvt_flat_kernel<<<1024, thr, 0, stream>>>(w_hh_b0, whhB0, GATES * 512);
  cvt_flat_kernel<<<1024, thr, 0, stream>>>(w_ih_f1, wihF1, GATES * 1024);
  cvt_flat_kernel<<<1024, thr, 0, stream>>>(w_hh_f1, whhF1, GATES * 512);
  cvt_flat_kernel<<<1024, thr, 0, stream>>>(w_ih_b1, wihB1, GATES * 1024);
  cvt_flat_kernel<<<1024, thr, 0, stream>>>(w_hh_b1, whhB1, GATES * 512);

  // layer 0: KA = 1024 -> LDS = 32*1024*2 + 32*512*4 = 128 KB
  size_t lds0 = (size_t)BATCH * (512 + HID) * sizeof(bf16) +
                (size_t)BATCH * HID * sizeof(float);
  lstm_scan_kernel<512, false><<<2, 1024, lds0, stream>>>(
      xbf, wihF0, b_ih_f0, whhF0, wihB0, b_ih_b0, whhB0, y0, nullptr);

  // layer 1: KA = 1536 -> LDS = 32*1536*2 + 32*512*4 = 160 KB
  size_t lds1 = (size_t)BATCH * (1024 + HID) * sizeof(bf16) +
                (size_t)BATCH * HID * sizeof(float);
  lstm_scan_kernel<1024, true><<<2, 1024, lds1, stream>>>(
      y0, wihF1, b_ih_f1, whhF1, wihB1, b_ih_b1, whhB1, nullptr,
      (float*)d_out);
}
